// GATNet_9723805958221
// MI455X (gfx1250) — matrix operand order repaired and checked by reading, not compiled, not run on hardware
//
#include <hip/hip_runtime.h>
#include <hip/hip_bf16.h>
#include <math.h>

// ---------------------------------------------------------------------------
// GATNet for MI455X (gfx1250, wave32).
//  - Dense projections: bf16 V_WMMA_F32_16X16X32_BF16, f32 accumulate.
//    One wave computes a 16 x (16*NT) output strip; K-loop is software-
//    pipelined (double-buffered A/B packs) so next-step loads overlap the
//    current step's NT WMMAs.
//  - Edge softmax/aggregation: L2-resident atomics (segment max via monotone
//    uint key + atomicMax, segment sums via global_atomic_add_f32); the
//    1/denom is deferred to the node-wise epilogue so only 2 edge passes run.
// ---------------------------------------------------------------------------

typedef __bf16 v16bf __attribute__((ext_vector_type(16)));
typedef __bf16 v8bf  __attribute__((ext_vector_type(8)));
typedef __bf16 v4bf  __attribute__((ext_vector_type(4)));
typedef float  v8f   __attribute__((ext_vector_type(8)));
typedef float  v4f   __attribute__((ext_vector_type(4)));

union BPack { v16bf v; v8bf h[2]; };

// ---------------- f32 -> bf16 conversion (vectorized) ----------------
__global__ void k_f32_to_bf16(const float* __restrict__ in,
                              __bf16* __restrict__ out, int n) {
  int i = (blockIdx.x * blockDim.x + threadIdx.x) * 4;
  if (i + 3 < n) {
    v4f f = *(const v4f*)(in + i);
    v4bf b;
    b.x = (__bf16)f.x; b.y = (__bf16)f.y; b.z = (__bf16)f.z; b.w = (__bf16)f.w;
    *(v4bf*)(out + i) = b;
  } else {
    for (; i < n; ++i) out[i] = (__bf16)in[i];
  }
}

// ---------------- WMMA GEMM: C[M,Nout] = A[M,128] * W[Nout,128]^T ----------
// One wave per 16x(16*NT) output strip; K = 128 in four 16x16x32 bf16 steps,
// double-buffered so step s+1 loads overlap step s WMMAs.
template <int NT>
__global__ void k_wmma_gemm(const __bf16* __restrict__ A,
                            const __bf16* __restrict__ W,
                            float* __restrict__ C, int M, int Nout) {
  const int tileM = blockIdx.x * 16;
  const int tileN = blockIdx.y * (16 * NT);
  const int lane  = threadIdx.x;      // 0..31 (wave32)
  const int r     = lane & 15;
  const int hi    = lane >> 4;        // 0 or 1

  int rA = tileM + r;                 // clamp for ragged M (reads only)
  if (rA >= M) rA = M - 1;
  const __bf16* arow = A + (size_t)rA * 128;
  const __bf16* wrow[NT];
#pragma unroll
  for (int t = 0; t < NT; t++)
    wrow[t] = W + (size_t)(tileN + t * 16 + r) * 128;

  v8f acc[NT];
#pragma unroll
  for (int t = 0; t < NT; t++) acc[t] = (v8f){};

  // A 16x32 (MxK): lanes 0-15 -> K {k0..k0+7, k0+16..k0+23},
  //                lanes 16-31 -> K {k0+8..k0+15, k0+24..k0+31}
  // B 32x16 (KxN): lane col n=r holds K same as A (k0+8*hi..+7,
  //                k0+16+8*hi..+7); W row-major [Nout,128]
  BPack a[2];
  BPack b[2][NT];
  auto load_step = [&](int buf, int k0) {
    a[buf].h[0] = *(const v8bf*)(arow + k0 + hi * 8);
    a[buf].h[1] = *(const v8bf*)(arow + k0 + 16 + hi * 8);
#pragma unroll
    for (int t = 0; t < NT; t++) {
      b[buf][t].h[0] = *(const v8bf*)(wrow[t] + k0 + hi * 8);
      b[buf][t].h[1] = *(const v8bf*)(wrow[t] + k0 + 16 + hi * 8);
    }
  };

  load_step(0, 0);
#pragma unroll
  for (int s = 0; s < 4; s++) {
    if (s + 1 < 4) load_step((s + 1) & 1, (s + 1) * 32);
#pragma unroll
    for (int t = 0; t < NT; t++)
      acc[t] = __builtin_amdgcn_wmma_f32_16x16x32_bf16(
          false, a[s & 1].v, false, b[s & 1][t].v, (short)0, acc[t],
          false, false);
  }

  // D layout: VGPR v -> row tileM + v + 8*hi, col tileN + t*16 + r
  if (tileM + 16 <= M) {
    // full tile: straight-line stores (wave-uniform branch, no exec churn)
    float* base = C + (size_t)(tileM + hi * 8) * Nout + r + tileN;
#pragma unroll
    for (int t = 0; t < NT; t++) {
      float* cp = base + t * 16;
#pragma unroll
      for (int v = 0; v < 8; v++) cp[(size_t)v * Nout] = acc[t][v];
    }
  } else {
#pragma unroll
    for (int t = 0; t < NT; t++)
#pragma unroll
      for (int v = 0; v < 8; v++) {
        int row = tileM + hi * 8 + v;
        if (row < M) C[(size_t)row * Nout + tileN + t * 16 + r] = acc[t][v];
      }
  }
}

// ---------------- attention scores: a[n,h] = <feat[n,h,:], att[h,:]> --------
__global__ void k_att_scores(const float* __restrict__ feat,
                             const float* __restrict__ att_src,
                             const float* __restrict__ att_dst,
                             float* __restrict__ asrc, float* __restrict__ adst,
                             int N, int H, int C) {
  int i = blockIdx.x * blockDim.x + threadIdx.x;
  if (i >= N * H) return;
  int n = i / H, hh = i % H;
  const float* fp = feat + (size_t)n * H * C + hh * C;
  float s = 0.f, d = 0.f;
  for (int c = 0; c < C; c++) {
    float v = fp[c];
    s = fmaf(v, att_src[hh * C + c], s);
    d = fmaf(v, att_dst[hh * C + c], d);
  }
  asrc[i] = s;
  adst[i] = d;
}

__device__ __forceinline__ unsigned float_to_key(float f) {
  unsigned b = __float_as_uint(f);
  return (b & 0x80000000u) ? ~b : (b | 0x80000000u);
}
__device__ __forceinline__ float key_to_float(unsigned k) {
  unsigned b = (k & 0x80000000u) ? (k & 0x7FFFFFFFu) : ~k;
  return __uint_as_float(b);
}

// ---------------- edge pass 1: segment max of leaky-relu score --------------
__global__ void k_edge_max(const int* __restrict__ ei, int E, int N,
                           const float* __restrict__ asrc,
                           const float* __restrict__ adst,
                           unsigned* __restrict__ mkey, int H) {
  int i = blockIdx.x * blockDim.x + threadIdx.x;
  int total = (E + N) * H;
  if (i >= total) return;
  int e = i / H, hh = i % H;
  int s, d;
  if (e < E) { s = ei[e]; d = ei[E + e]; } else { s = d = e - E; }
  float x = asrc[s * H + hh] + adst[d * H + hh];
  x = x > 0.f ? x : 0.2f * x;                   // leaky relu
  atomicMax(&mkey[d * H + hh], float_to_key(x));
}

// ---------------- edge pass 2: accumulate exp-weighted features -------------
// One wave per edge (grid-stride). C must be 32 (one lane per channel).
__global__ void k_edge_agg(const int* __restrict__ ei, int E, int N,
                           const float* __restrict__ asrc,
                           const float* __restrict__ adst,
                           const unsigned* __restrict__ mkey,
                           const float* __restrict__ feat,
                           float* __restrict__ agg, float* __restrict__ denom,
                           int H, int C) {
  const int lane = threadIdx.x & 31;
  const int wave = (blockIdx.x * blockDim.x + threadIdx.x) >> 5;
  const int nw   = (gridDim.x * blockDim.x) >> 5;
  const int nE   = E + N;
  for (int e = wave; e < nE; e += nw) {
    // prefetch next iteration's edge index entries (global_prefetch_b8)
    int en = e + nw;
    if (en < E) {
      __builtin_prefetch(&ei[en], 0, 3);
      __builtin_prefetch(&ei[E + en], 0, 3);
    }
    int s, d;
    if (e < E) { s = ei[e]; d = ei[E + e]; } else { s = d = e - E; }
    float exv = 0.f;
    if (lane < H) {
      float x = asrc[s * H + lane] + adst[d * H + lane];
      x = x > 0.f ? x : 0.2f * x;
      float m = key_to_float(mkey[d * H + lane]);
      exv = __expf(x - m);
      atomicAdd(&denom[d * H + lane], exv);
    }
    for (int hh = 0; hh < H; hh++) {
      float ex = __shfl(exv, hh, 32);
      float v = feat[(size_t)s * H * C + hh * C + lane] * ex;
      atomicAdd(&agg[(size_t)d * H * C + hh * C + lane], v);
    }
  }
}

// ---------------- layer-0 epilogue: /denom + bias + skip + ELU --------------
__global__ void k_finalize0(const float* __restrict__ agg,
                            const float* __restrict__ denom,
                            const float* __restrict__ skip,
                            const float* __restrict__ bias,
                            const float* __restrict__ linb,
                            float* __restrict__ hout, __bf16* __restrict__ hb,
                            int N, int H, int C) {
  int i = blockIdx.x * blockDim.x + threadIdx.x;
  int HC = H * C;
  if (i >= N * HC) return;
  int n = i / HC, hc = i % HC, hh = hc / C;
  float v = agg[i] / (denom[n * H + hh] + 1e-16f) + bias[hc] + skip[i] + linb[hc];
  v = v > 0.f ? v : expm1f(v);                  // ELU (alpha = 1)
  hout[i] = v;
  hb[i] = (__bf16)v;
}

// ---------------- layer-1 epilogue: /denom + bias + skip --------------------
__global__ void k_finalize1(const float* __restrict__ agg,
                            const float* __restrict__ denom,
                            const float* __restrict__ skip,
                            const float* __restrict__ bias,
                            const float* __restrict__ linb,
                            float* __restrict__ out, int N, int C) {
  int i = blockIdx.x * blockDim.x + threadIdx.x;
  if (i >= N * C) return;
  int n = i / C, c = i % C;
  out[i] = agg[i] / (denom[n] + 1e-16f) + bias[c] + skip[i] + linb[c];
}

// ---------------------------------------------------------------------------
extern "C" void kernel_launch(void* const* d_in, const int* in_sizes, int n_in,
                              void* d_out, int out_size, void* d_ws, size_t ws_size,
                              hipStream_t stream) {
  const float* x        = (const float*)d_in[0];
  const int*   ei       = (const int*)d_in[1];
  const float* W0       = (const float*)d_in[2];
  const float* att_src0 = (const float*)d_in[3];
  const float* att_dst0 = (const float*)d_in[4];
  const float* bias0    = (const float*)d_in[5];
  const float* linW0    = (const float*)d_in[6];
  const float* linb0    = (const float*)d_in[7];
  const float* W1       = (const float*)d_in[8];
  const float* att_src1 = (const float*)d_in[9];
  const float* att_dst1 = (const float*)d_in[10];
  const float* bias1    = (const float*)d_in[11];
  const float* linW1    = (const float*)d_in[12];
  const float* linb1    = (const float*)d_in[13];

  const int IN = 128, H0 = 4, C0 = 32, H1 = 1, C1 = 32;
  const int HC0 = H0 * C0;                 // 128
  const int N = in_sizes[0] / IN;
  const int E = in_sizes[1] / 2;
  const int NE = E + N;

  // ---- workspace carve-out (256B aligned) ----
  size_t off = 0;
  auto alloc = [&](size_t bytes) -> void* {
    void* p = (char*)d_ws + off;
    off += (bytes + 255) & ~(size_t)255;
    return p;
  };
  __bf16* xb      = (__bf16*)alloc((size_t)N * IN * sizeof(__bf16)); // reused as hb
  __bf16* W0b     = (__bf16*)alloc((size_t)HC0 * IN * sizeof(__bf16));
  __bf16* linW0b  = (__bf16*)alloc((size_t)HC0 * IN * sizeof(__bf16));
  __bf16* W1b     = (__bf16*)alloc((size_t)C1 * HC0 * sizeof(__bf16));
  __bf16* linW1b  = (__bf16*)alloc((size_t)C1 * HC0 * sizeof(__bf16));
  float*  h0      = (float*)alloc((size_t)N * HC0 * sizeof(float));  // reused as h
  float*  skip0   = (float*)alloc((size_t)N * HC0 * sizeof(float));
  float*  agg0    = (float*)alloc((size_t)N * HC0 * sizeof(float));
  float*  asrc0   = (float*)alloc((size_t)N * H0 * sizeof(float));
  float*  adst0   = (float*)alloc((size_t)N * H0 * sizeof(float));
  unsigned* mkey0 = (unsigned*)alloc((size_t)N * H0 * sizeof(unsigned));
  float*  denom0  = (float*)alloc((size_t)N * H0 * sizeof(float));
  float*  g1      = (float*)alloc((size_t)N * C1 * sizeof(float));
  float*  skip1   = (float*)alloc((size_t)N * C1 * sizeof(float));
  float*  agg1    = (float*)alloc((size_t)N * C1 * sizeof(float));
  float*  asrc1   = (float*)alloc((size_t)N * H1 * sizeof(float));
  float*  adst1   = (float*)alloc((size_t)N * H1 * sizeof(float));
  unsigned* mkey1 = (unsigned*)alloc((size_t)N * H1 * sizeof(unsigned));
  float*  denom1  = (float*)alloc((size_t)N * H1 * sizeof(float));
  (void)ws_size; (void)n_in;

  const int TB = 256;
  auto cdiv = [](int a, int b) { return (a + b - 1) / b; };

  // ---- convert inputs to bf16 ----
  k_f32_to_bf16<<<cdiv(N * IN, TB * 4), TB, 0, stream>>>(x, xb, N * IN);
  k_f32_to_bf16<<<cdiv(HC0 * IN, TB * 4), TB, 0, stream>>>(W0, W0b, HC0 * IN);
  k_f32_to_bf16<<<cdiv(HC0 * IN, TB * 4), TB, 0, stream>>>(linW0, linW0b, HC0 * IN);
  k_f32_to_bf16<<<cdiv(C1 * HC0, TB * 4), TB, 0, stream>>>(W1, W1b, C1 * HC0);
  k_f32_to_bf16<<<cdiv(C1 * HC0, TB * 4), TB, 0, stream>>>(linW1, linW1b, C1 * HC0);

  // ---- layer 0 dense projections (WMMA, 16x64 strip per wave) ----
  dim3 g0(cdiv(N, 16), HC0 / 64);
  k_wmma_gemm<4><<<g0, 32, 0, stream>>>(xb, W0b, h0, N, HC0);
  k_wmma_gemm<4><<<g0, 32, 0, stream>>>(xb, linW0b, skip0, N, HC0);

  // ---- layer 0 attention ----
  k_att_scores<<<cdiv(N * H0, TB), TB, 0, stream>>>(h0, att_src0, att_dst0,
                                                    asrc0, adst0, N, H0, C0);
  hipMemsetAsync(mkey0, 0, (size_t)N * H0 * sizeof(unsigned), stream);
  hipMemsetAsync(denom0, 0, (size_t)N * H0 * sizeof(float), stream);
  hipMemsetAsync(agg0, 0, (size_t)N * HC0 * sizeof(float), stream);
  k_edge_max<<<cdiv(NE * H0, TB), TB, 0, stream>>>(ei, E, N, asrc0, adst0,
                                                   mkey0, H0);
  k_edge_agg<<<2048, TB, 0, stream>>>(ei, E, N, asrc0, adst0, mkey0, h0,
                                      agg0, denom0, H0, C0);
  // epilogue: h (f32) overwrites h0; hb (bf16) overwrites xb
  __bf16* hb = xb;
  k_finalize0<<<cdiv(N * HC0, TB), TB, 0, stream>>>(agg0, denom0, skip0, bias0,
                                                    linb0, h0, hb, N, H0, C0);

  // ---- layer 1 dense projections (WMMA, 16x32 strip per wave) ----
  dim3 g1d(cdiv(N, 16), C1 / 32);
  k_wmma_gemm<2><<<g1d, 32, 0, stream>>>(hb, W1b, g1, N, C1);
  k_wmma_gemm<2><<<g1d, 32, 0, stream>>>(hb, linW1b, skip1, N, C1);

  // ---- layer 1 attention ----
  k_att_scores<<<cdiv(N * H1, TB), TB, 0, stream>>>(g1, att_src1, att_dst1,
                                                    asrc1, adst1, N, H1, C1);
  hipMemsetAsync(mkey1, 0, (size_t)N * H1 * sizeof(unsigned), stream);
  hipMemsetAsync(denom1, 0, (size_t)N * H1 * sizeof(float), stream);
  hipMemsetAsync(agg1, 0, (size_t)N * C1 * sizeof(float), stream);
  k_edge_max<<<cdiv(NE * H1, TB), TB, 0, stream>>>(ei, E, N, asrc1, adst1,
                                                   mkey1, H1);
  k_edge_agg<<<2048, TB, 0, stream>>>(ei, E, N, asrc1, adst1, mkey1, g1,
                                      agg1, denom1, H1, C1);
  k_finalize1<<<cdiv(N * C1, TB), TB, 0, stream>>>(agg1, denom1, skip1, bias1,
                                                   linb1, (float*)d_out, N, C1);
}
